// GRUModel_47725676593785
// MI455X (gfx1250) — matrix operand order repaired and checked by reading, not compiled, not run on hardware
//
#include <hip/hip_runtime.h>
#include <hip/hip_bf16.h>

typedef __attribute__((ext_vector_type(16))) _Float16 v16h;
typedef __attribute__((ext_vector_type(8)))  _Float16 v8h;
typedef __attribute__((ext_vector_type(8)))  float    v8f;
typedef __attribute__((ext_vector_type(4)))  unsigned int u32x4;
typedef __attribute__((ext_vector_type(8)))  int i32x8;
typedef __attribute__((ext_vector_type(4)))  int i32x4;

#define T_SEQ 18
#define HID   64
#define RPB   32      // batch rows per block
#define NTHR  128     // 4 waves: 2 fwd + 2 bwd

// dynamic LDS layout (bytes)
#define OFF_BGRU 0        // [2][12][2][512] f16  = 49152 B  (W_hh B-fragments)
#define OFF_BFC1 49152    // [2][2][2][512]  f16  = 8192 B   (fc1_w B-fragments)
#define OFF_XS   57344    // [32][18][2]     f32  = 4608 B
#define OFF_HST  61952    // [4][16][64]     f16  = 8192 B   (per-wave h staging)
#define OFF_FC1P 70144    // [2][32][18][24] f32  = 110592 B (fc1 partials, fwd/bwd)
#define SMEM_BYTES 180736

__device__ __forceinline__ v16h cat8(v8h lo, v8h hi) {
  v16h r;
#pragma unroll
  for (int i = 0; i < 8; ++i) { r[i] = lo[i]; r[i + 8] = hi[i]; }
  return r;
}

__device__ __forceinline__ v8f wmma16(v16h a, v16h b, v8f c) {
  return __builtin_amdgcn_wmma_f32_16x16x32_f16(false, a, false, b, (short)0, c,
                                                false, false);
}

// branch-free activations: single v_exp_f32 + v_rcp_f32 (co-execute with WMMA)
__device__ __forceinline__ float fast_sigmoid(float x) {
  return __builtin_amdgcn_rcpf(
      1.f + __builtin_amdgcn_exp2f(-1.4426950408889634f * x));
}
__device__ __forceinline__ float fast_tanh(float x) {
  // tanh(x) = 2/(1+exp(-2x)) - 1
  float s = __builtin_amdgcn_rcpf(
      1.f + __builtin_amdgcn_exp2f(-2.8853900817779268f * x));
  return fmaf(2.f, s, -1.f);
}

// D = A(16x64,f16) * Bfrag-pair(64x16,f16) + 0   (two chained K=32 WMMAs)
__device__ __forceinline__ v8f gemm64(const _Float16* fragbase, int lane,
                                      v16h A0, v16h A1) {
  const _Float16* b0 = fragbase + lane * 16;
  const _Float16* b1 = fragbase + 512 + lane * 16;
  v16h B0 = cat8(*(const v8h*)b0, *(const v8h*)(b0 + 8));
  v16h B1 = cat8(*(const v8h*)b1, *(const v8h*)(b1 + 8));
  v8f c = {0.f, 0.f, 0.f, 0.f, 0.f, 0.f, 0.f, 0.f};
  c = wmma16(A0, B0, c);
  c = wmma16(A1, B1, c);
  return c;
}

__global__ void __launch_bounds__(NTHR)
bigru_fused(const float* __restrict__ x,
            const float* __restrict__ wih_f, const float* __restrict__ whh_f,
            const float* __restrict__ bih_f, const float* __restrict__ bhh_f,
            const float* __restrict__ wih_b, const float* __restrict__ whh_b,
            const float* __restrict__ bih_b, const float* __restrict__ bhh_b,
            const float* __restrict__ fc1w, const float* __restrict__ fc1b,
            const float* __restrict__ fc2w, const float* __restrict__ fc2b,
            float* __restrict__ out) {
  extern __shared__ unsigned char smem[];
  _Float16* Bgru = (_Float16*)(smem + OFF_BGRU);
  _Float16* Bfc1 = (_Float16*)(smem + OFF_BFC1);
  float*    xs   = (float*)(smem + OFF_XS);
  _Float16* hst  = (_Float16*)(smem + OFF_HST);
  float*    fc1p = (float*)(smem + OFF_FC1P);

  const int tid    = threadIdx.x;
  const int wave   = tid >> 5;
  const int lane   = tid & 31;
  const int lanelo = lane & 15;
  const int half   = lane >> 4;
  const int dir    = wave >> 1;   // 0 = forward, 1 = backward
  const int rg     = wave & 1;    // row-group within block
  const int row0   = rg * 16;
  const long gbase = (long)blockIdx.x * RPB;

  // ---- TDM: async DMA of this block's x tile (1152 f32) into LDS ----------
  // 1-D tensor: data_size=4B, tensor_dim0 = tile_dim0 = 1152, stride = 1152.
  if (wave == 0) {
    const float* xg = x + gbase * (T_SEQ * 2);
    unsigned long long ga = (unsigned long long)(uintptr_t)xg;
    unsigned int ldsoff   = (unsigned int)(uintptr_t)xs;  // low32 = LDS offset
    u32x4 g0 = {1u,                                  // count=1, user mode
                ldsoff,                              // lds_addr
                (unsigned int)ga,                    // global_addr[31:0]
                (unsigned int)((ga >> 32) & 0x01FFFFFFu) | 0x80000000u}; // +type=2
    i32x8 g1 = {(int)(2u << 16),        // data_size = 4B
                (int)(1152u << 16),     // tensor_dim0[15:0]
                (int)(1u << 16),        // tensor_dim0[31:16]=0 | tensor_dim1=1
                (int)(1152u << 16),     // tensor_dim1 hi=0 | tile_dim0=1152
                1,                      // tile_dim1=1, tile_dim2=0
                1152,                   // tensor_dim0_stride[31:0]
                0, 0};
    i32x4 z4 = {0, 0, 0, 0};
#if defined(__clang_major__) && (__clang_major__ >= 23)
    i32x8 z8 = {0, 0, 0, 0, 0, 0, 0, 0};
    __builtin_amdgcn_tensor_load_to_lds(g0, g1, z4, z4, z8, 0);
#else
    __builtin_amdgcn_tensor_load_to_lds(g0, g1, z4, z4, 0);
#endif
  }

  // ---- pre-swizzle W_hh (both dirs) into WMMA B-fragment order ----
  // element e of lane ln holds W[k = ks*32 + (ln>=16)*16 + e][n = jt*16 + (ln&15)]
  for (int idx = tid; idx < 2 * 12 * 2 * 512; idx += NTHR) {
    int d  = idx / 12288;
    int r  = idx - d * 12288;
    int jt = r >> 10;
    int r2 = r & 1023;
    int ks = r2 >> 9;
    int r3 = r2 & 511;
    int ln = r3 >> 4;
    int e  = r3 & 15;
    int n  = jt * 16 + (ln & 15);
    int k  = ks * 32 + (ln >> 4) * 16 + e;
    const float* whh = d ? whh_b : whh_f;
    Bgru[idx] = (_Float16)whh[n * HID + k];
  }
  // ---- pre-swizzle fc1_w into B-fragment order (split by direction halves) ----
  for (int idx = tid; idx < 2 * 2 * 2 * 512; idx += NTHR) {
    int d  = idx >> 11;
    int r  = idx & 2047;
    int ct = r >> 10;
    int r2 = r & 1023;
    int ks = r2 >> 9;
    int r3 = r2 & 511;
    int ln = r3 >> 4;
    int e  = r3 & 15;
    int c  = ct * 16 + (ln & 15);
    int k  = ks * 32 + (ln >> 4) * 16 + e;
    Bfc1[idx] = (c < 24) ? (_Float16)fc1w[c * 128 + d * 64 + k] : (_Float16)0.f;
  }
  // ---- zero h staging + fc1 partial accumulators ----
  for (int i = tid; i < 4 * 16 * HID; i += NTHR) hst[i] = (_Float16)0.f;
  for (int i = tid; i < 2 * RPB * T_SEQ * 24; i += NTHR) fc1p[i] = 0.f;

  // ---- per-lane input-projection constants (I=2 -> just 2 FMAs per gate) ----
  const float* wih = dir ? wih_b : wih_f;
  const float* bih = dir ? bih_b : bih_f;
  const float* bhh = dir ? bhh_b : bhh_f;
  float wA[12], wB[12], bias[12];
#pragma unroll
  for (int g = 0; g < 12; ++g) {
    int n   = g * 16 + lanelo;
    wA[g]   = wih[n * 2 + 0];
    wB[g]   = wih[n * 2 + 1];
    bias[g] = bih[n] + bhh[n];
  }
  float bias1[2];
#pragma unroll
  for (int ct = 0; ct < 2; ++ct) {
    int c = ct * 16 + lanelo;
    bias1[ct] = (dir == 0 && c < 24) ? fc1b[c] : 0.f;  // add fc1 bias exactly once
  }
  if (wave == 0) __builtin_amdgcn_s_wait_tensorcnt(0);  // x tile landed in LDS
  __syncthreads();

  _Float16* myh = hst + wave * 16 * HID;  // this wave's 16x64 h staging tile
  float ho[4][8];                          // h_old in C-fragment layout
#pragma unroll
  for (int j = 0; j < 4; ++j)
#pragma unroll
    for (int v = 0; v < 8; ++v) ho[j][v] = 0.f;

  // =================== recurrent loop (sequential in T) ===================
  for (int s = 0; s < T_SEQ; ++s) {
    const int t = dir ? (T_SEQ - 1 - s) : s;

    float x0[8], x1[8];
#pragma unroll
    for (int v = 0; v < 8; ++v) {
      int m = v + half * 8;
      x0[v] = xs[(row0 + m) * (T_SEQ * 2) + t * 2 + 0];
      x1[v] = xs[(row0 + m) * (T_SEQ * 2) + t * 2 + 1];
    }

    // h_old as 16-bit A fragments (contiguous layout: K = half*16 + e)
    const _Float16* ha = myh + lanelo * HID;
    v16h A0 = cat8(*(const v8h*)(ha + 0 + half * 16),
                   *(const v8h*)(ha + 8 + half * 16));
    v16h A1 = cat8(*(const v8h*)(ha + 32 + half * 16),
                   *(const v8h*)(ha + 40 + half * 16));

    float hn[4][8];
#pragma unroll
    for (int j = 0; j < 4; ++j) {  // hidden column tiles of 16
      v8f aR = gemm64(Bgru + (dir * 12 + j) * 1024, lane, A0, A1);
      v8f aZ = gemm64(Bgru + (dir * 12 + 4 + j) * 1024, lane, A0, A1);
      v8f aN = gemm64(Bgru + (dir * 12 + 8 + j) * 1024, lane, A0, A1);
#pragma unroll
      for (int v = 0; v < 8; ++v) {
        float ir = fmaf(x0[v], wA[j],     fmaf(x1[v], wB[j],     bias[j]))     + aR[v];
        float iz = fmaf(x0[v], wA[4 + j], fmaf(x1[v], wB[4 + j], bias[4 + j])) + aZ[v];
        float in_= fmaf(x0[v], wA[8 + j], fmaf(x1[v], wB[8 + j], bias[8 + j]));
        float r = fast_sigmoid(ir);
        float z = fast_sigmoid(iz);
        float n = fast_tanh(in_ + r * aN[v]);
        hn[j][v] = (1.f - z) * n + z * ho[j][v];
      }
    }

    // C-layout -> LDS staging (transposes for next A-fragment reload)
#pragma unroll
    for (int j = 0; j < 4; ++j)
#pragma unroll
      for (int v = 0; v < 8; ++v) {
        int m = v + half * 8;
        myh[m * HID + j * 16 + lanelo] = (_Float16)hn[j][v];
        ho[j][v] = hn[j][v];
      }

    // ---- fused fc1 partial: this direction's 64-wide K slice ----
    v16h N0 = cat8(*(const v8h*)(ha + 0 + half * 16),
                   *(const v8h*)(ha + 8 + half * 16));
    v16h N1 = cat8(*(const v8h*)(ha + 32 + half * 16),
                   *(const v8h*)(ha + 40 + half * 16));
#pragma unroll
    for (int ct = 0; ct < 2; ++ct) {
      v8f acc;
#pragma unroll
      for (int v = 0; v < 8; ++v) acc[v] = bias1[ct];
      const _Float16* fb = Bfc1 + (dir * 2 + ct) * 1024;
      v16h B0 = cat8(*(const v8h*)(fb + lane * 16),
                     *(const v8h*)(fb + lane * 16 + 8));
      v16h B1 = cat8(*(const v8h*)(fb + 512 + lane * 16),
                     *(const v8h*)(fb + 512 + lane * 16 + 8));
      acc = wmma16(N0, B0, acc);
      acc = wmma16(N1, B1, acc);
      int c = ct * 16 + lanelo;
      if (c < 24) {
#pragma unroll
        for (int v = 0; v < 8; ++v) {
          int m = v + half * 8;
          fc1p[((dir * RPB + row0 + m) * T_SEQ + t) * 24 + c] += acc[v];
        }
      }
    }
  }
  __syncthreads();

  // =================== fc2 epilogue (raw reshape (B,T,24)->(B,24,18)) ========
  const float b20 = fc2b[0], b21 = fc2b[1];
  for (int p = tid; p < RPB * 24; p += NTHR) {
    int row = p / 24;
    int i   = p - row * 24;
    float s0 = b20, s1 = b21;
#pragma unroll
    for (int j = 0; j < T_SEQ; ++j) {
      int f  = i * T_SEQ + j;
      int tt = f / 24;
      int cc = f - tt * 24;
      float v = fc1p[(row * T_SEQ + tt) * 24 + cc] +
                fc1p[((RPB + row) * T_SEQ + tt) * 24 + cc];
      s0 = fmaf(v, fc2w[j], s0);
      s1 = fmaf(v, fc2w[T_SEQ + j], s1);
    }
    long b = gbase + row;
    out[b * 48 + i * 2 + 0] = s0;
    out[b * 48 + i * 2 + 1] = s1;
  }
}

extern "C" void kernel_launch(void* const* d_in, const int* in_sizes, int n_in,
                              void* d_out, int out_size, void* d_ws, size_t ws_size,
                              hipStream_t stream) {
  (void)in_sizes; (void)n_in; (void)d_ws; (void)ws_size; (void)out_size;
  const float* x     = (const float*)d_in[0];
  const float* wih_f = (const float*)d_in[1];
  const float* whh_f = (const float*)d_in[2];
  const float* bih_f = (const float*)d_in[3];
  const float* bhh_f = (const float*)d_in[4];
  const float* wih_b = (const float*)d_in[5];
  const float* whh_b = (const float*)d_in[6];
  const float* bih_b = (const float*)d_in[7];
  const float* bhh_b = (const float*)d_in[8];
  const float* fc1w  = (const float*)d_in[9];
  const float* fc1b  = (const float*)d_in[10];
  const float* fc2w  = (const float*)d_in[11];
  const float* fc2b  = (const float*)d_in[12];
  float* out = (float*)d_out;

  const int blocks = 32768 / RPB;  // 1024
  bigru_fused<<<blocks, NTHR, SMEM_BYTES, stream>>>(
      x, wih_f, whh_f, bih_f, bhh_f, wih_b, whh_b, bih_b, bhh_b,
      fc1w, fc1b, fc2w, fc2b, out);
}
